// TextGenerationModel_90881507983835
// MI455X (gfx1250) — compile-verified, matrix operand order checked by reading, not run on hardware
//
#include <hip/hip_runtime.h>
#include <hip/hip_bf16.h>
#include <stdint.h>
#include <stddef.h>

// ---------------------------------------------------------------------------
// 2-layer LSTM + FC + log_softmax, fully fused per 16-row batch tile.
// B=256, S=1024, V=256, H=256, gates G=4H=1024.
// Grid: 16 workgroups (one per 16-row WMMA M-tile), 512 threads = 16 wave32s.
// Wave w owns hidden columns [16w, 16w+16); cell state c stays in VGPRs for
// all 1024 timesteps; h tiles live in LDS; weights stream bf16 from L2.
// x tiles are DMA'd one step ahead via global_load_async_to_lds_b128.
// ---------------------------------------------------------------------------

typedef __attribute__((ext_vector_type(16))) __bf16 v16bf;
typedef __attribute__((ext_vector_type(8)))  float  v8f;

#define B_TOT 256
#define S_LEN 1024
#define V_DIM 256
#define H_DIM 256
#define G_DIM 1024
#define KDIM  256      // K for every GEMM in this net (V == H == 256)
#define BT    16       // batch rows per workgroup (one WMMA M-tile)
#define NTHR  512      // 16 wave32s
#define LDST  264      // LDS row stride in halves: 528B, 16B aligned,
                       // 132 dwords/row -> A-frag lanes hit 16 distinct banks
#define XBUF  (BT * 256)   // one fp32 x staging buffer (16KB)

union Frag16 {
    v16bf v;
    uint4 q[2];
};

__device__ __forceinline__ unsigned short f2bf(float f) {
    unsigned int u = __float_as_uint(f);
    u += 0x7FFFu + ((u >> 16) & 1u);          // round-to-nearest-even
    return (unsigned short)(u >> 16);
}

// CDNA5 ISA 7.12.2: for 16-bit A (16x32) a lane holds two contiguous 8-elem
// K-runs: K in [k0+8*half, +8) and [k0+16+8*half, +8). The B (32x16) layout
// mirrors this with lane = N column. Both become 2x 16-byte loads.
__device__ __forceinline__ Frag16 load_frag(const unsigned short* row_base,
                                            int k0, int half) {
    Frag16 f;
    const unsigned short* p = row_base + k0 + 8 * half;
    f.q[0] = *reinterpret_cast<const uint4*>(p);
    f.q[1] = *reinterpret_cast<const uint4*>(p + 16);
    return f;
}

__device__ __forceinline__ v8f wmma_bf16(const Frag16& a, const Frag16& b, v8f c) {
    return __builtin_amdgcn_wmma_f32_16x16x32_bf16(
        /*neg_a=*/false, a.v, /*neg_b=*/false, b.v,
        /*c_mod=*/(short)0, c, /*reuse_a=*/false, /*reuse_b=*/false);
}

// ---- fast transcendentals: v_rcp_f32 / v_tanh_f32 instead of full division --
__device__ __forceinline__ float rcp_fast(float x) {
#if __has_builtin(__builtin_amdgcn_rcpf)
    return __builtin_amdgcn_rcpf(x);
#else
    return 1.0f / x;
#endif
}
__device__ __forceinline__ float sigf(float x) {
    return rcp_fast(1.0f + __expf(-x));
}
__device__ __forceinline__ float tanh_fast(float x) {
#if __has_builtin(__builtin_amdgcn_tanhf)
    return __builtin_amdgcn_tanhf(x);
#else
    const float e = __expf(2.0f * x);
    return (e - 1.0f) * rcp_fast(e + 1.0f);
#endif
}

// ---- CDNA5 async global->LDS copy (ASYNCcnt-tracked DMA, no VGPR bounce) ---
__device__ __forceinline__ void async_copy_b128(unsigned lds_byte_off,
                                                const void* gaddr) {
    asm volatile("global_load_async_to_lds_b128 %0, %1, off"
                 :: "v"(lds_byte_off), "v"(gaddr)
                 : "memory");
}
__device__ __forceinline__ void wait_async0() {
    asm volatile("s_wait_asynccnt 0x0" ::: "memory");
}
// Generic pointer into the LDS aperture: hardware maps addr[31:0] to the LDS
// offset (ISA 10.2 aperture rules), so truncation yields the DMA destination.
__device__ __forceinline__ unsigned lds_off(const void* p) {
    return (unsigned)(uintptr_t)p;
}

// One LSTM timestep for one layer. a_in / h_ls are LDS tiles [BT x LDST] bf16;
// wih/whh are bf16 weight matrices [G_DIM x KDIM] row-major in global (L2).
// cst = per-lane cell state (one 16-col tile). Reads h_ls as A, then (after a
// barrier) overwrites it with the new h for this wave's columns.
// A fragments are double-buffered so DS latency overlaps the WMMAs.
__device__ __forceinline__ void lstm_step(
    const unsigned short* a_in,
    unsigned short*       h_ls,
    const unsigned short* __restrict__ wih,
    const unsigned short* __restrict__ whh,
    const float bias[4], v8f* cst,
    int nloc, int half, int hbase)
{
    v8f acc[4];
#pragma unroll
    for (int g = 0; g < 4; ++g)
#pragma unroll
        for (int r = 0; r < 8; ++r) acc[g][r] = bias[g];

    Frag16 axp[2], ahp[2];
    axp[0] = load_frag(a_in + nloc * LDST, 0, half);
    ahp[0] = load_frag(h_ls + nloc * LDST, 0, half);
#pragma unroll
    for (int kt = 0; kt < 8; ++kt) {            // K = 256 = 8 x 32
        const int k0  = kt * 32;
        const int cur = kt & 1, nxt = cur ^ 1;
        if (kt < 7) {                           // prefetch next A fragments
            axp[nxt] = load_frag(a_in + nloc * LDST, k0 + 32, half);
            ahp[nxt] = load_frag(h_ls + nloc * LDST, k0 + 32, half);
        }
#pragma unroll
        for (int g = 0; g < 4; ++g) {           // gate order: i, f, g, o
            const int ncol = g * H_DIM + hbase + nloc;
            Frag16 bi = load_frag(wih + (size_t)ncol * KDIM, k0, half);
            Frag16 bh = load_frag(whh + (size_t)ncol * KDIM, k0, half);
            acc[g] = wmma_bf16(axp[cur], bi, acc[g]);
            acc[g] = wmma_bf16(ahp[cur], bh, acc[g]);
        }
    }
    __syncthreads();                            // all waves done reading h_ls

    // Gate nonlinearities + state update directly on accumulator layout:
    // lane holds column hbase+nloc, rows m = r + 8*half.
#pragma unroll
    for (int r = 0; r < 8; ++r) {
        float iv = sigf(acc[0][r]);
        float fv = sigf(acc[1][r]);
        float gv = tanh_fast(acc[2][r]);
        float ov = sigf(acc[3][r]);
        float c  = fv * (*cst)[r] + iv * gv;
        (*cst)[r] = c;
        h_ls[(r + 8 * half) * LDST + hbase + nloc] = f2bf(ov * tanh_fast(c));
    }
    __syncthreads();                            // new h visible to all waves
}

__global__ __launch_bounds__(NTHR, 1)
void lstm2_fused_kernel(
    const float*          __restrict__ x,       // [B, S, V] fp32
    const unsigned short* __restrict__ wih0,    // bf16 [G, V]
    const unsigned short* __restrict__ whh0,    // bf16 [G, H]
    const unsigned short* __restrict__ wih1,    // bf16 [G, H]
    const unsigned short* __restrict__ whh1,    // bf16 [G, H]
    const unsigned short* __restrict__ fcw,     // bf16 [V, H]
    const float*          __restrict__ bias0,   // [G] = b_ih0 + b_hh0
    const float*          __restrict__ bias1,   // [G]
    const float*          __restrict__ fcb,     // [V]
    float*                __restrict__ out)     // [B, S, V] log-probs
{
    __shared__ __align__(16) unsigned short xls [BT * LDST];  // layer0 input
    __shared__ __align__(16) unsigned short h0ls[BT * LDST];  // layer0 h
    __shared__ __align__(16) unsigned short h1ls[BT * LDST];  // layer1 h
    __shared__ __align__(16) float          lgls[BT * LDST];  // logits fp32
    __shared__ __align__(16) float          xf32[2 * XBUF];   // async x staging

    const int tid   = threadIdx.x;
    const int wave  = tid >> 5;
    const int lane  = tid & 31;
    const int nloc  = lane & 15;
    const int half  = lane >> 4;
    const int hbase = wave * 16;                // this wave's 16 columns
    const int b0    = blockIdx.x * BT;

    // Each thread owns exactly 8 fp32 x elements per tile: row xr, cols xc..xc+7.
    const int xr = tid >> 5;
    const int xc = (tid & 31) * 8;

    for (int i = tid; i < BT * LDST; i += NTHR) { h0ls[i] = 0; h1ls[i] = 0; }

    // Time-invariant per-lane bias values (fp32, become the WMMA C init).
    float b0v[4], b1v[4];
#pragma unroll
    for (int g = 0; g < 4; ++g) {
        b0v[g] = bias0[g * H_DIM + hbase + nloc];
        b1v[g] = bias1[g * H_DIM + hbase + nloc];
    }
    const float fbv = fcb[hbase + nloc];

    v8f c0s, c1s;
#pragma unroll
    for (int r = 0; r < 8; ++r) { c0s[r] = 0.f; c1s[r] = 0.f; }

    // Kick off async DMA of the t=0 x tile into buffer 0.
    {
        const unsigned dst = lds_off(&xf32[xr * 256 + xc]);
        const float*   gp  = &x[((size_t)(b0 + xr) * S_LEN) * V_DIM + xc];
        async_copy_b128(dst,      gp);
        async_copy_b128(dst + 16, gp + 4);
    }
    __syncthreads();

    for (int t = 0; t < S_LEN; ++t) {
        // ---- x tile t has been DMA'd into xf32[t&1]; convert own 8 elems ----
        wait_async0();
        {
            const float* src = &xf32[(t & 1) * XBUF + xr * 256 + xc];
            union { uint4 q; unsigned short h[8]; } o;
#pragma unroll
            for (int i = 0; i < 8; ++i) o.h[i] = f2bf(src[i]);
            *reinterpret_cast<uint4*>(&xls[xr * LDST + xc]) = o.q;
        }
        // ---- launch DMA for tile t+1 into the other buffer ----
        if (t + 1 < S_LEN) {
            const unsigned dst = lds_off(&xf32[((t + 1) & 1) * XBUF + xr * 256 + xc]);
            const float*   gp  = &x[((size_t)(b0 + xr) * S_LEN + (t + 1)) * V_DIM + xc];
            async_copy_b128(dst,      gp);
            async_copy_b128(dst + 16, gp + 4);
        }
        // keep L2 warm two steps ahead (global_prefetch_b8)
        if (t + 2 < S_LEN)
            __builtin_prefetch(&x[((size_t)(b0 + xr) * S_LEN + (t + 2)) * V_DIM + xc], 0, 1);
        __syncthreads();

        // ---- layer 0 then layer 1 (fused, h1 never leaves LDS) ----
        lstm_step(xls,  h0ls, wih0, whh0, b0v, &c0s, nloc, half, hbase);
        lstm_step(h0ls, h1ls, wih1, whh1, b1v, &c1s, nloc, half, hbase);

        // ---- FC projection: logits = h2 @ fc_w^T + fc_b ----
        {
            v8f lacc;
#pragma unroll
            for (int r = 0; r < 8; ++r) lacc[r] = fbv;
            Frag16 ap[2];
            ap[0] = load_frag(h1ls + nloc * LDST, 0, half);
#pragma unroll
            for (int kt = 0; kt < 8; ++kt) {
                const int cur = kt & 1, nxt = cur ^ 1;
                if (kt < 7)
                    ap[nxt] = load_frag(h1ls + nloc * LDST, (kt + 1) * 32, half);
                Frag16 bw = load_frag(fcw + (size_t)(hbase + nloc) * H_DIM,
                                      kt * 32, half);
                lacc = wmma_bf16(ap[cur], bw, lacc);
            }
#pragma unroll
            for (int r = 0; r < 8; ++r)
                lgls[(r + 8 * half) * LDST + hbase + nloc] = lacc[r];
        }
        __syncthreads();

        // ---- row-wise log_softmax over V=256; wave handles row m=wave ----
        {
            const int m = wave;
            float v[8];
            float mx = -3.402823466e38f;
#pragma unroll
            for (int i = 0; i < 8; ++i) {
                v[i] = lgls[m * LDST + lane + 32 * i];
                mx = fmaxf(mx, v[i]);
            }
#pragma unroll
            for (int off = 16; off > 0; off >>= 1)
                mx = fmaxf(mx, __shfl_xor(mx, off, 32));
            float s = 0.f;
#pragma unroll
            for (int i = 0; i < 8; ++i) s += __expf(v[i] - mx);
#pragma unroll
            for (int off = 16; off > 0; off >>= 1)
                s += __shfl_xor(s, off, 32);
            const float lse = mx + __logf(s);
            float* op = out + ((size_t)(b0 + m) * S_LEN + t) * V_DIM;
#pragma unroll
            for (int i = 0; i < 8; ++i) op[lane + 32 * i] = v[i] - lse;
        }
        __syncthreads();   // protect lgls vs next iteration's FC writes
    }
}

// ---------------------------------------------------------------------------
// Prologue: fp32 -> bf16 weight conversion, bias folding (b_ih + b_hh).
// ---------------------------------------------------------------------------
__global__ void f32_to_bf16_kernel(const float* __restrict__ in,
                                   unsigned short* __restrict__ o, int n) {
    const int i = blockIdx.x * blockDim.x + threadIdx.x;
    if (i < n) o[i] = f2bf(in[i]);
}

__global__ void bias_sum_kernel(const float* __restrict__ a,
                                const float* __restrict__ b,
                                float* __restrict__ o, int n) {
    const int i = blockIdx.x * blockDim.x + threadIdx.x;
    if (i < n) o[i] = a[i] + b[i];
}

extern "C" void kernel_launch(void* const* d_in, const int* in_sizes, int n_in,
                              void* d_out, int out_size, void* d_ws, size_t ws_size,
                              hipStream_t stream) {
    const float* x     = (const float*)d_in[0];
    const float* wih0f = (const float*)d_in[1];
    const float* whh0f = (const float*)d_in[2];
    const float* bih0  = (const float*)d_in[3];
    const float* bhh0  = (const float*)d_in[4];
    const float* wih1f = (const float*)d_in[5];
    const float* whh1f = (const float*)d_in[6];
    const float* bih1  = (const float*)d_in[7];
    const float* bhh1  = (const float*)d_in[8];
    const float* fcwf  = (const float*)d_in[9];
    const float* fcb   = (const float*)d_in[10];
    float* out = (float*)d_out;

    // Workspace layout (bf16 weights + folded biases), ~2.24 MB total.
    char* ws = (char*)d_ws;
    const size_t NW = (size_t)G_DIM * KDIM;           // 262144 elems per LSTM mat
    unsigned short* wih0b = (unsigned short*)(ws);
    unsigned short* whh0b = (unsigned short*)(ws + 2 * NW);
    unsigned short* wih1b = (unsigned short*)(ws + 4 * NW);
    unsigned short* whh1b = (unsigned short*)(ws + 6 * NW);
    unsigned short* fcwb  = (unsigned short*)(ws + 8 * NW);
    float* bias0 = (float*)(ws + 8 * NW + 2 * (size_t)V_DIM * H_DIM);
    float* bias1 = bias0 + G_DIM;

    const int nmat = (int)NW;                         // 262144
    const int nfc  = V_DIM * H_DIM;                   // 65536
    f32_to_bf16_kernel<<<(nmat + 255) / 256, 256, 0, stream>>>(wih0f, wih0b, nmat);
    f32_to_bf16_kernel<<<(nmat + 255) / 256, 256, 0, stream>>>(whh0f, whh0b, nmat);
    f32_to_bf16_kernel<<<(nmat + 255) / 256, 256, 0, stream>>>(wih1f, wih1b, nmat);
    f32_to_bf16_kernel<<<(nmat + 255) / 256, 256, 0, stream>>>(whh1f, whh1b, nmat);
    f32_to_bf16_kernel<<<(nfc + 255) / 256, 256, 0, stream>>>(fcwf, fcwb, nfc);
    bias_sum_kernel<<<(G_DIM + 255) / 256, 256, 0, stream>>>(bih0, bhh0, bias0, G_DIM);
    bias_sum_kernel<<<(G_DIM + 255) / 256, 256, 0, stream>>>(bih1, bhh1, bias1, G_DIM);

    lstm2_fused_kernel<<<B_TOT / BT, NTHR, 0, stream>>>(
        x, wih0b, whh0b, wih1b, whh1b, fcwb, bias0, bias1, fcb, out);
}
